// PointerGenerator_64828236366287
// MI455X (gfx1250) — compile-verified, matrix operand order checked by reading, not run on hardware
//
#include <hip/hip_runtime.h>
#include <hip/hip_bf16.h>
#include <math.h>

typedef _Float16 f16;
typedef __attribute__((ext_vector_type(16))) _Float16 v16h;
typedef __attribute__((ext_vector_type(8)))  float    v8f;

#define B_   16
#define L_   400
#define T_   50
#define H_   256
#define E_   128
#define V_   32000
#define H2_  512    // 2H (decoder hidden)
#define G4_  1024   // 4H (encoder gates / feature width)
#define G8_  2048   // 8H (decoder gates)

// k index inside a 32-wide K slab for the 16-bit WMMA A/B fragment layout
// (ISA 7.12.2): lanes 0-15 -> K slabs {0..7,16..23}, lanes 16-31 -> {8..15,24..31}
__device__ __forceinline__ int frag_k(int khalf, int e) {
  return (e < 8) ? (khalf * 8 + e) : (16 + khalf * 8 + (e - 8));
}
__device__ __forceinline__ float sigmoidf_(float x) { return 1.0f / (1.0f + __expf(-x)); }

// ---------------- one-time packing ----------------
// Pack row-major f32 W[N,K] into per-fragment layout:
//   out[((nt*(K/32)+ks)*32 + lane)*16 + e] = W[nt*16 + lane%16, ks*32 + frag_k(lane/16,e)]
__global__ void pack_w_kernel(const float* __restrict__ W, f16* __restrict__ out, int N, int K) {
  int KS = K >> 5;
  int total = (N >> 4) * KS * 512;
  for (int idx = blockIdx.x * blockDim.x + threadIdx.x; idx < total; idx += gridDim.x * blockDim.x) {
    int e = idx & 15, lane = (idx >> 4) & 31, tile = idx >> 9;
    int ks = tile % KS, nt = tile / KS;
    int n = nt * 16 + (lane & 15);
    int k = ks * 32 + frag_k(lane >> 4, e);
    out[idx] = (f16)W[n * K + k];
  }
}

// Pack embedded source tokens x_emb[b,l,:] (optionally time-reversed) as A fragments,
// rows = b*L + l, K = E.
__global__ void pack_x_kernel(const float* __restrict__ emb, const int* __restrict__ src,
                              f16* __restrict__ out, int rev) {
  int total = (B_ * L_ / 16) * (E_ / 32) * 512;  // 819200
  for (int idx = blockIdx.x * blockDim.x + threadIdx.x; idx < total; idx += gridDim.x * blockDim.x) {
    int e = idx & 15, lane = (idx >> 4) & 31, tile = idx >> 9;
    int ks = tile & 3, mt = tile >> 2;  // E/32 = 4 k-steps
    int row = mt * 16 + (lane & 15);
    int b = row / L_, l = row % L_;
    int ls = rev ? (L_ - 1 - l) : l;
    int k = ks * 32 + frag_k(lane >> 4, e);
    out[idx] = (f16)emb[(long)src[b * L_ + ls] * E_ + k];
  }
}

// ---------------- encoder input GEMM: Xpre[6400,1024] = X @ Wih.T + b ----------------
__global__ void __launch_bounds__(256) enc_pre_gemm_kernel(const f16* __restrict__ pX,
                                                           const f16* __restrict__ pW,
                                                           const float* __restrict__ bias,
                                                           float* __restrict__ Xpre) {
  int wave = threadIdx.x >> 5, lane = threadIdx.x & 31;
  int tile = blockIdx.x * 8 + wave;     // 400*64 = 25600 tiles
  int mt = tile >> 6, nt = tile & 63;
  int nl = lane & 15, mh = lane >> 4;
  float bv = bias[nt * 16 + nl];
  v8f acc;
#pragma unroll
  for (int r = 0; r < 8; ++r) acc[r] = bv;
#pragma unroll
  for (int ks = 0; ks < 4; ++ks) {
    v16h a = *(const v16h*)(pX + (((long)mt * 4 + ks) * 32 + lane) * 16);
    v16h b = *(const v16h*)(pW + (((long)nt * 4 + ks) * 32 + lane) * 16);
    acc = __builtin_amdgcn_wmma_f32_16x16x32_f16(false, a, false, b, (short)0, acc, false, false);
  }
#pragma unroll
  for (int r = 0; r < 8; ++r) {
    int m = r + 8 * mh;
    Xpre[((long)(mt * 16 + m)) * G4_ + nt * 16 + nl] = acc[r];
  }
}

// ---------------- encoder recurrence (persistent, 1 block per direction) ----------------
// 16 waves; wave jt owns j-range [jt*16, jt*16+16) and computes the i,f,g,o tiles for it,
// so the gate nonlinearity needs no LDS staging. h kept in LDS already in A-fragment order.
__global__ void __launch_bounds__(512) enc_recurrent_kernel(
    const float* __restrict__ XpreF, const float* __restrict__ XpreB,
    const f16* __restrict__ pWhhF, const f16* __restrict__ pWhhB,
    float* __restrict__ memory, float* __restrict__ hstate, float* __restrict__ cstate) {
  __shared__ __align__(32) f16 hpk[8 * 512];   // 8 k-steps of A fragments (K=256)
  __shared__ float cst[16 * 256];
  int dir = blockIdx.x;
  const float* Xpre = dir ? XpreB : XpreF;
  const f16* pW = dir ? pWhhB : pWhhF;
  int tid = threadIdx.x;
  for (int i = tid; i < 8 * 512; i += 512) hpk[i] = (f16)0.0f;
  for (int i = tid; i < 16 * 256; i += 512) cst[i] = 0.0f;
  __syncthreads();
  int jt = tid >> 5, lane = tid & 31;
  int nl = lane & 15, mh = lane >> 4;
  for (int t = 0; t < L_; ++t) {
    v8f acc[4];
#pragma unroll
    for (int g = 0; g < 4; ++g) {
      int n = g * H_ + jt * 16 + nl;
#pragma unroll
      for (int r = 0; r < 8; ++r) {
        int m = r + 8 * mh;
        acc[g][r] = Xpre[((long)m * L_ + t) * G4_ + n];
      }
    }
#pragma unroll
    for (int ks = 0; ks < 8; ++ks) {
      v16h a = *(const v16h*)(hpk + ks * 512 + lane * 16);
#pragma unroll
      for (int g = 0; g < 4; ++g) {
        int nt = g * 16 + jt;
        v16h b = *(const v16h*)(pW + (((long)nt * 8 + ks) * 32 + lane) * 16);
        acc[g] = __builtin_amdgcn_wmma_f32_16x16x32_f16(false, a, false, b, (short)0, acc[g], false, false);
      }
    }
    __syncthreads();  // everyone done reading hpk
#pragma unroll
    for (int r = 0; r < 8; ++r) {
      int m = r + 8 * mh;
      int j = jt * 16 + nl;
      float iv = acc[0][r], fv = acc[1][r], gv = acc[2][r], ov = acc[3][r];
      float c2 = sigmoidf_(fv) * cst[m * H_ + j] + sigmoidf_(iv) * tanhf(gv);
      float h2 = sigmoidf_(ov) * tanhf(c2);
      cst[m * H_ + j] = c2;
      // write h2 back into A-fragment layout for next step
      int ks2 = j >> 5, r2 = j & 31;
      int kh = (r2 >> 3) & 1, e2 = (r2 & 7) + ((r2 >= 16) ? 8 : 0);
      hpk[ks2 * 512 + (kh * 16 + m) * 16 + e2] = (f16)h2;
      int l = dir ? (L_ - 1 - t) : t;
      memory[((long)m * L_ + l) * H2_ + dir * H_ + j] = h2;
      if (t == L_ - 1) {
        hstate[m * H2_ + dir * H_ + j] = h2;
        cstate[m * H2_ + dir * H_ + j] = c2;
      }
    }
    __syncthreads();  // hpk/cst updates visible before next step
  }
}

// ---------------- decoder LSTM step (1 block, 32 waves) ----------------
__global__ void __launch_bounds__(1024) dec_lstm_kernel(
    const float* __restrict__ emb, const int* __restrict__ tokbuf,
    float* __restrict__ hstate, float* __restrict__ cstate,
    const f16* __restrict__ pDWih, const f16* __restrict__ pDWhh,
    const float* __restrict__ dec_b, float* __restrict__ feature) {
  __shared__ __align__(32) f16 sA[20 * 512];  // K = 128 (emb) + 512 (h) = 640 -> 20 k-steps
  int tid = threadIdx.x;
  for (int idx = tid; idx < 20 * 512; idx += 1024) {
    int e = idx & 15, lane2 = (idx >> 4) & 31, ks = idx >> 9;
    int m = lane2 & 15, kk = frag_k(lane2 >> 4, e);
    float v;
    if (ks < 4) v = emb[(long)tokbuf[m] * E_ + ks * 32 + kk];
    else        v = hstate[m * H2_ + (ks - 4) * 32 + kk];
    sA[idx] = (f16)v;
  }
  __syncthreads();
  int jt = tid >> 5, lane = tid & 31;
  int nl = lane & 15, mh = lane >> 4;
  v8f acc[4];
#pragma unroll
  for (int g = 0; g < 4; ++g) {
    float bv = dec_b[g * H2_ + jt * 16 + nl];
#pragma unroll
    for (int r = 0; r < 8; ++r) acc[g][r] = bv;
  }
  for (int ks = 0; ks < 20; ++ks) {
    v16h a = *(const v16h*)(sA + ks * 512 + lane * 16);
#pragma unroll
    for (int g = 0; g < 4; ++g) {
      int nt = g * 32 + jt;  // (g*512 + jt*16)/16
      const f16* bp = (ks < 4) ? (pDWih + (((long)nt * 4 + ks) * 32 + lane) * 16)
                               : (pDWhh + (((long)nt * 16 + (ks - 4)) * 32 + lane) * 16);
      v16h b = *(const v16h*)bp;
      acc[g] = __builtin_amdgcn_wmma_f32_16x16x32_f16(false, a, false, b, (short)0, acc[g], false, false);
    }
  }
#pragma unroll
  for (int r = 0; r < 8; ++r) {
    int m = r + 8 * mh, j = jt * 16 + nl;
    float iv = acc[0][r], fv = acc[1][r], gv = acc[2][r], ov = acc[3][r];
    float c2 = sigmoidf_(fv) * cstate[m * H2_ + j] + sigmoidf_(iv) * tanhf(gv);
    float h2 = sigmoidf_(ov) * tanhf(c2);
    cstate[m * H2_ + j] = c2;
    hstate[m * H2_ + j] = h2;
    feature[m * G4_ + j] = h2;  // feature = [h2 | ctx]
  }
}

// ---------------- attention: scores + softmax ----------------
__global__ void __launch_bounds__(256) attn_scores_kernel(const float* __restrict__ feature,
                                                          const float* __restrict__ memory,
                                                          float* __restrict__ att) {
  __shared__ float sH[512];
  __shared__ float sS[L_];
  __shared__ float red[256];
  int b = blockIdx.x, tid = threadIdx.x;
  for (int i = tid; i < 512; i += 256) sH[i] = feature[b * G4_ + i];
  __syncthreads();
  for (int l = tid; l < L_; l += 256) {
    const float* mp = memory + ((long)b * L_ + l) * H2_;
    float s = 0.f;
    for (int d = 0; d < H2_; ++d) s += sH[d] * mp[d];
    sS[l] = s;
  }
  __syncthreads();
  float lm = -3.4e38f;
  for (int l = tid; l < L_; l += 256) lm = fmaxf(lm, sS[l]);
  red[tid] = lm; __syncthreads();
  for (int s = 128; s > 0; s >>= 1) { if (tid < s) red[tid] = fmaxf(red[tid], red[tid + s]); __syncthreads(); }
  float mx = red[0]; __syncthreads();
  float ls = 0.f;
  for (int l = tid; l < L_; l += 256) ls += __expf(sS[l] - mx);
  red[tid] = ls; __syncthreads();
  for (int s = 128; s > 0; s >>= 1) { if (tid < s) red[tid] += red[tid + s]; __syncthreads(); }
  float inv = 1.f / red[0];
  for (int l = tid; l < L_; l += 256) att[b * L_ + l] = __expf(sS[l] - mx) * inv;
}

// ---------------- ctx = att @ memory, into feature[:,512:1024] ----------------
__global__ void __launch_bounds__(512) attn_ctx_kernel(const float* __restrict__ att,
                                                       const float* __restrict__ memory,
                                                       float* __restrict__ feature) {
  __shared__ float sA[L_];
  int b = blockIdx.x, tid = threadIdx.x;  // 512 threads == H2_
  for (int i = tid; i < L_; i += 512) sA[i] = att[b * L_ + i];
  __syncthreads();
  float acc = 0.f;
  for (int l = 0; l < L_; ++l) acc += sA[l] * memory[((long)b * L_ + l) * H2_ + tid];
  feature[b * G4_ + H2_ + tid] = acc;
}

__global__ void zero_kernel(float* __restrict__ p, int n) {
  for (int i = blockIdx.x * blockDim.x + threadIdx.x; i < n; i += gridDim.x * blockDim.x) p[i] = 0.f;
}

__global__ void scatter_kernel(const int* __restrict__ src, const float* __restrict__ att,
                               float* __restrict__ ptrbuf) {
  int i = blockIdx.x * blockDim.x + threadIdx.x;
  if (i < B_ * L_) {
    int b = i / L_;
    atomicAdd(&ptrbuf[(long)b * V_ + src[i]], att[i]);
  }
}

// ---------------- pack feature into A fragments + pgen head ----------------
__global__ void __launch_bounds__(512) pack_feat_pgen_kernel(
    const float* __restrict__ feature, const float* __restrict__ emb,
    const int* __restrict__ tokbuf, const float* __restrict__ Wpg,
    const float* __restrict__ bpg, f16* __restrict__ featA, float* __restrict__ pgen) {
  int tid = threadIdx.x;
  for (int idx = tid; idx < 32 * 512; idx += 512) {  // K=1024 -> 32 k-steps
    int e = idx & 15, lane2 = (idx >> 4) & 31, ks = idx >> 9;
    int m = lane2 & 15;
    int k = ks * 32 + frag_k(lane2 >> 4, e);
    featA[idx] = (f16)feature[m * G4_ + k];
  }
  // pgen: one wave per batch row; pgen_feat = [ctx | h2 | ans_emb]
  int b = tid >> 5, lane = tid & 31;
  float s = 0.f;
  for (int d = lane; d < G4_ + E_; d += 32) {
    float x;
    if (d < H2_)      x = feature[b * G4_ + H2_ + d];          // ctx
    else if (d < G4_) x = feature[b * G4_ + (d - H2_)];        // h2
    else              x = emb[(long)tokbuf[b] * E_ + (d - G4_)];  // ans_emb
    s += x * Wpg[d];
  }
#pragma unroll
  for (int off = 16; off > 0; off >>= 1) s += __shfl_down(s, off, 32);
  if (lane == 0) pgen[b] = sigmoidf_(s + bpg[0]);
}

// ---------------- vocab projection: logits[16,32000] = feature @ Wpro.T + bpro ----------------
// 4 N-tiles per wave -> 500 waves in flight (vs 250 at 8/wave): better latency hiding
// for the L2-resident 65MB Wpro stream while still amortizing A-fragment loads 4x.
#define VG_NT 4
__global__ void __launch_bounds__(256) vocab_gemm_kernel(
    const f16* __restrict__ featA, const f16* __restrict__ pWpro,
    const float* __restrict__ bpro, float* __restrict__ logits) {
  int wave = threadIdx.x >> 5, lane = threadIdx.x & 31;
  int wg = blockIdx.x * 8 + wave;
  if (wg >= (V_ / 16) / VG_NT) return;  // 2000 n-tiles / 4 per wave = 500 waves
  int tile0 = wg * VG_NT;
  int nl = lane & 15, mh = lane >> 4;
  v8f acc[VG_NT];
#pragma unroll
  for (int t = 0; t < VG_NT; ++t)
#pragma unroll
    for (int r = 0; r < 8; ++r) acc[t][r] = 0.f;
  for (int ks = 0; ks < 32; ++ks) {
    v16h a = *(const v16h*)(featA + ks * 512 + lane * 16);  // broadcast, cache-hit
#pragma unroll
    for (int t = 0; t < VG_NT; ++t) {
      long nt = tile0 + t;
      v16h b = *(const v16h*)(pWpro + ((nt * 32 + ks) * 32 + lane) * 16);
      acc[t] = __builtin_amdgcn_wmma_f32_16x16x32_f16(false, a, false, b, (short)0, acc[t], false, false);
    }
  }
#pragma unroll
  for (int t = 0; t < VG_NT; ++t) {
    int n = (tile0 + t) * 16 + nl;
    float bv = bpro[n];
#pragma unroll
    for (int r = 0; r < 8; ++r) {
      int m = r + 8 * mh;
      logits[(long)m * V_ + n] = acc[t][r] + bv;
    }
  }
}

// ---------------- per-row max / sum-exp over vocab ----------------
__global__ void __launch_bounds__(256) rowred_kernel(const float* __restrict__ logits,
                                                     float* __restrict__ mx, float* __restrict__ Zs) {
  __shared__ float red[256];
  int b = blockIdx.x, tid = threadIdx.x;
  float lm = -3.4e38f;
  for (int v = tid; v < V_; v += 256) lm = fmaxf(lm, logits[(long)b * V_ + v]);
  red[tid] = lm; __syncthreads();
  for (int s = 128; s > 0; s >>= 1) { if (tid < s) red[tid] = fmaxf(red[tid], red[tid + s]); __syncthreads(); }
  float m = red[0]; __syncthreads();
  float ls = 0.f;
  for (int v = tid; v < V_; v += 256) ls += __expf(logits[(long)b * V_ + v] - m);
  red[tid] = ls; __syncthreads();
  for (int s = 128; s > 0; s >>= 1) { if (tid < s) red[tid] += red[tid + s]; __syncthreads(); }
  if (tid == 0) { mx[b] = m; Zs[b] = red[0]; }
}

// ---------------- final mix + argmax; writes outputs and next token ----------------
__global__ void __launch_bounds__(256) final_kernel(
    const float* __restrict__ logits, const float* __restrict__ ptrbuf,
    const float* __restrict__ pgen, const float* __restrict__ mx, const float* __restrict__ Zs,
    int* __restrict__ tokbuf, int* __restrict__ out_tok, float* __restrict__ out_val, int t) {
  __shared__ float rv[256];
  __shared__ int ri[256];
  int b = blockIdx.x, tid = threadIdx.x;
  float p = pgen[b], m = mx[b], iz = 1.f / Zs[b];
  float best = -3.4e38f; int bi = 0;
  for (int v = tid; v < V_; v += 256) {
    float f = p * __expf(logits[(long)b * V_ + v] - m) * iz + (1.f - p) * ptrbuf[(long)b * V_ + v] + 1e-8f;
    if (f > best) { best = f; bi = v; }
  }
  rv[tid] = best; ri[tid] = bi; __syncthreads();
  for (int s = 128; s > 0; s >>= 1) {
    if (tid < s) {
      if (rv[tid + s] > rv[tid] || (rv[tid + s] == rv[tid] && ri[tid + s] < ri[tid])) {
        rv[tid] = rv[tid + s]; ri[tid] = ri[tid + s];
      }
    }
    __syncthreads();
  }
  if (tid == 0) {
    tokbuf[b] = ri[0];
    out_tok[b * T_ + t] = ri[0];
    out_val[b * T_ + t] = logf(rv[0]);
  }
}

__global__ void init_tok_kernel(const int* __restrict__ start, int* __restrict__ tokbuf) {
  if (threadIdx.x < B_) tokbuf[threadIdx.x] = start[0];
}

extern "C" void kernel_launch(void* const* d_in, const int* in_sizes, int n_in,
                              void* d_out, int out_size, void* d_ws, size_t ws_size,
                              hipStream_t stream) {
  const int*   src      = (const int*)d_in[0];
  const int*   start    = (const int*)d_in[3];
  const float* emb      = (const float*)d_in[4];
  const float* eWihF    = (const float*)d_in[5];
  const float* eWhhF    = (const float*)d_in[6];
  const float* ebF      = (const float*)d_in[7];
  const float* eWihB    = (const float*)d_in[8];
  const float* eWhhB    = (const float*)d_in[9];
  const float* ebB      = (const float*)d_in[10];
  const float* dWih     = (const float*)d_in[11];
  const float* dWhh     = (const float*)d_in[12];
  const float* db       = (const float*)d_in[13];
  const float* Wpro     = (const float*)d_in[14];
  const float* bpro     = (const float*)d_in[15];
  const float* Wpg      = (const float*)d_in[16];
  const float* bpg      = (const float*)d_in[17];

  size_t off = 0;
  auto alloc = [&](size_t bytes) -> void* {
    void* p = (char*)d_ws + off;
    off += (bytes + 255) & ~(size_t)255;
    return p;
  };
  f16*   pXf    = (f16*)alloc((size_t)B_ * L_ * E_ * 2);
  f16*   pXb    = (f16*)alloc((size_t)B_ * L_ * E_ * 2);
  f16*   pWihF  = (f16*)alloc((size_t)G4_ * E_ * 2);
  f16*   pWihB  = (f16*)alloc((size_t)G4_ * E_ * 2);
  f16*   pWhhF  = (f16*)alloc((size_t)G4_ * H_ * 2);
  f16*   pWhhB  = (f16*)alloc((size_t)G4_ * H_ * 2);
  f16*   pDWih  = (f16*)alloc((size_t)G8_ * E_ * 2);
  f16*   pDWhh  = (f16*)alloc((size_t)G8_ * H2_ * 2);
  f16*   pWpro  = (f16*)alloc((size_t)V_ * G4_ * 2);
  float* XpreF  = (float*)alloc((size_t)B_ * L_ * G4_ * 4);
  float* XpreB  = (float*)alloc((size_t)B_ * L_ * G4_ * 4);
  float* memory = (float*)alloc((size_t)B_ * L_ * H2_ * 4);
  float* hstate = (float*)alloc((size_t)B_ * H2_ * 4);
  float* cstate = (float*)alloc((size_t)B_ * H2_ * 4);
  float* feature= (float*)alloc((size_t)B_ * G4_ * 4);
  f16*   featA  = (f16*)alloc((size_t)32 * 512 * 2);
  float* att    = (float*)alloc((size_t)B_ * L_ * 4);
  float* logits = (float*)alloc((size_t)B_ * V_ * 4);
  float* ptrbuf = (float*)alloc((size_t)B_ * V_ * 4);
  float* mx     = (float*)alloc(64 * 4);
  float* Zs     = (float*)alloc(64 * 4);
  float* pgen   = (float*)alloc(64 * 4);
  int*   tokbuf = (int*)alloc(64 * 4);

  int* out_tok   = (int*)d_out;
  float* out_val = (float*)d_out + B_ * T_;

  auto gsz = [](long total) { long g = (total + 255) / 256; return (int)(g > 8192 ? 8192 : g); };

  // one-time packs
  pack_w_kernel<<<gsz((long)(G4_/16)*(E_/32)*512), 256, 0, stream>>>(eWihF, pWihF, G4_, E_);
  pack_w_kernel<<<gsz((long)(G4_/16)*(E_/32)*512), 256, 0, stream>>>(eWihB, pWihB, G4_, E_);
  pack_w_kernel<<<gsz((long)(G4_/16)*(H_/32)*512), 256, 0, stream>>>(eWhhF, pWhhF, G4_, H_);
  pack_w_kernel<<<gsz((long)(G4_/16)*(H_/32)*512), 256, 0, stream>>>(eWhhB, pWhhB, G4_, H_);
  pack_w_kernel<<<gsz((long)(G8_/16)*(E_/32)*512), 256, 0, stream>>>(dWih, pDWih, G8_, E_);
  pack_w_kernel<<<gsz((long)(G8_/16)*(H2_/32)*512), 256, 0, stream>>>(dWhh, pDWhh, G8_, H2_);
  pack_w_kernel<<<gsz((long)(V_/16)*(G4_/32)*512), 256, 0, stream>>>(Wpro, pWpro, V_, G4_);
  pack_x_kernel<<<gsz((long)(B_*L_/16)*(E_/32)*512), 256, 0, stream>>>(emb, src, pXf, 0);
  pack_x_kernel<<<gsz((long)(B_*L_/16)*(E_/32)*512), 256, 0, stream>>>(emb, src, pXb, 1);

  // encoder
  enc_pre_gemm_kernel<<<(B_*L_/16)*(G4_/16)/8, 256, 0, stream>>>(pXf, pWihF, ebF, XpreF);
  enc_pre_gemm_kernel<<<(B_*L_/16)*(G4_/16)/8, 256, 0, stream>>>(pXb, pWihB, ebB, XpreB);
  enc_recurrent_kernel<<<2, 512, 0, stream>>>(XpreF, XpreB, pWhhF, pWhhB, memory, hstate, cstate);

  // decoder
  init_tok_kernel<<<1, 32, 0, stream>>>(start, tokbuf);
  const int vg_waves = (V_ / 16) / VG_NT;                 // 500
  const int vg_blocks = (vg_waves + 7) / 8;               // 63
  for (int t = 0; t < T_; ++t) {
    dec_lstm_kernel<<<1, 1024, 0, stream>>>(emb, tokbuf, hstate, cstate, pDWih, pDWhh, db, feature);
    attn_scores_kernel<<<B_, 256, 0, stream>>>(feature, memory, att);
    attn_ctx_kernel<<<B_, 512, 0, stream>>>(att, memory, feature);
    zero_kernel<<<2000, 256, 0, stream>>>(ptrbuf, B_ * V_);
    scatter_kernel<<<(B_ * L_ + 255) / 256, 256, 0, stream>>>(src, att, ptrbuf);
    pack_feat_pgen_kernel<<<1, 512, 0, stream>>>(feature, emb, tokbuf, Wpg, bpg, featA, pgen);
    vocab_gemm_kernel<<<vg_blocks, 256, 0, stream>>>(featA, pWpro, bpro, logits);
    rowred_kernel<<<B_, 256, 0, stream>>>(logits, mx, Zs);
    final_kernel<<<B_, 256, 0, stream>>>(logits, ptrbuf, pgen, mx, Zs, tokbuf, out_tok, out_val, t);
  }
}
